// SparseMoELanguageModel_7086696039162
// MI455X (gfx1250) — compile-verified, matrix operand order checked by reading, not run on hardware
//
#include <hip/hip_runtime.h>
#include <hip/hip_bf16.h>
#include <math.h>

#define Bsz  4
#define Tsz  1024
#define Dsz  768
#define Hn   12
#define Ln   4
#define En   8
#define CAPn 1024
#define Ntok (Bsz*Tsz)

typedef __attribute__((ext_vector_type(16))) __bf16 v16bf;
typedef __attribute__((ext_vector_type(8)))  float  v8f;

enum { F_BIAS=1, F_RELU=2, F_GATHER=4, F_SCATTER=8, F_ACCUM=16 };

__device__ __forceinline__ unsigned short f2bf(float f){
  union { float f; unsigned u; } a; a.f = f;
  unsigned r = a.u + 0x7FFFu + ((a.u >> 16) & 1u);   // round-to-nearest-even
  return (unsigned short)(r >> 16);
}

// Load a 16x32 bf16 WMMA A/B fragment from an LDS tile with row stride STRIDE
// (in ushorts). ISA layout: lanes 0-15 hold row/col = lane, K = {0..7,16..23};
// lanes 16-31 hold K = {8..15,24..31}.  Two 16-byte ds loads per lane.
template<int STRIDE>
__device__ __forceinline__ v16bf ldfrag(const unsigned short* base, int rc, int kofs, int lane){
  const unsigned short* p = base + (rc + (lane & 15)) * STRIDE + kofs + ((lane >> 4) << 3);
  union { v16bf v; uint4 q[2]; } u;
  u.q[0] = *(const uint4*)(p);
  u.q[1] = *(const uint4*)(p + 16);
  return u.v;
}

__device__ __forceinline__ v8f wmma_bf16(v16bf a, v16bf b, v8f c){
  return __builtin_amdgcn_wmma_f32_16x16x32_bf16(false, a, false, b, (short)0, c, false, false);
}

// ---------------------------------------------------------------------------
// Generic bf16 WMMA GEMM:  C[M,N] (fp32) = op(A[M,K] @ W[K,N] + bias)
// Block tile 256x128x32, 256 threads = 8 waves in 4x2; each wave owns a 64x64
// sub-tile (4x4 accumulators -> 16 WMMAs per 8 fragment loads per K-step).
// Register-staged ping-pong LDS: next K-tile global loads are issued before
// the current tile's WMMA burst; one barrier per K-step.
// M%256==0, N%128==0, K%32==0 (true for every GEMM here).
// ---------------------------------------------------------------------------
#define BM 256
#define BN 128
#define BK 32
#define LDT 40   // padded LDS row pitch in ushorts (80B: 16B aligned, conflict-free)

__launch_bounds__(256)
__global__ void gemm_bf16(const float* __restrict__ A, const float* __restrict__ W,
                          const float* __restrict__ bias, float* __restrict__ C,
                          int M, int N, int Kd,
                          const int* __restrict__ gidx,    // gather rows of A
                          const int* __restrict__ sidx,    // scatter rows of C
                          const float* __restrict__ srow,  // per-row scale for scatter
                          int flags)
{
  __shared__ __align__(16) unsigned short As[2][BM * LDT];
  __shared__ __align__(16) unsigned short Bs[2][BN * LDT];   // stored [n][k]
  const int tid  = threadIdx.x;
  const int lane = tid & 31, wid = tid >> 5;
  const int bm = blockIdx.y * BM, bn = blockIdx.x * BN;
  const int wm = (wid >> 1) * 64;   // wave row offset: 0/64/128/192
  const int wn = (wid & 1) * 64;    // wave col offset: 0/64

  v8f acc[4][4];
  #pragma unroll
  for (int mi = 0; mi < 4; ++mi)
    #pragma unroll
    for (int ni = 0; ni < 4; ++ni)
      acc[mi][ni] = (v8f){0,0,0,0,0,0,0,0};

  float4 pa[8], pb[4];   // register staging for the next K-tile

  auto loadTile = [&](int k0){
    #pragma unroll
    for (int i = 0; i < 8; ++i) {            // A: 256x32 fp32 = 2048 float4
      int q = tid + 256 * i;
      int r = q >> 3, c4 = (q & 7) * 4;
      int gr = bm + r;
      int ar = (flags & F_GATHER) ? gidx[gr] : gr;
      pa[i] = *(const float4*)(A + (size_t)ar * Kd + k0 + c4);
    }
    #pragma unroll
    for (int i = 0; i < 4; ++i) {            // B: 32x128 fp32 = 1024 float4
      int q = tid + 256 * i;
      int kr = q >> 5, c4 = (q & 31) * 4;
      pb[i] = *(const float4*)(W + (size_t)(k0 + kr) * N + bn + c4);
    }
  };
  auto stageTile = [&](int buf){
    #pragma unroll
    for (int i = 0; i < 8; ++i) {
      int q = tid + 256 * i;
      int r = q >> 3, c4 = (q & 7) * 4;
      unsigned short* d = &As[buf][r * LDT + c4];
      d[0] = f2bf(pa[i].x); d[1] = f2bf(pa[i].y); d[2] = f2bf(pa[i].z); d[3] = f2bf(pa[i].w);
    }
    #pragma unroll
    for (int i = 0; i < 4; ++i) {
      int q = tid + 256 * i;
      int kr = q >> 5, c4 = (q & 31) * 4;
      Bs[buf][(c4 + 0) * LDT + kr] = f2bf(pb[i].x);
      Bs[buf][(c4 + 1) * LDT + kr] = f2bf(pb[i].y);
      Bs[buf][(c4 + 2) * LDT + kr] = f2bf(pb[i].z);
      Bs[buf][(c4 + 3) * LDT + kr] = f2bf(pb[i].w);
    }
  };

  const int KT = Kd / BK;
  loadTile(0);
  stageTile(0);
  __syncthreads();

  for (int kt = 0; kt < KT; ++kt) {
    const int buf = kt & 1;
    if (kt + 1 < KT) loadTile((kt + 1) * BK);   // overlap global loads with WMMA

    v16bf af[4], bf4[4];
    #pragma unroll
    for (int mi = 0; mi < 4; ++mi) af[mi]  = ldfrag<LDT>(As[buf], wm + mi * 16, 0, lane);
    #pragma unroll
    for (int ni = 0; ni < 4; ++ni) bf4[ni] = ldfrag<LDT>(Bs[buf], wn + ni * 16, 0, lane);
    #pragma unroll
    for (int mi = 0; mi < 4; ++mi)
      #pragma unroll
      for (int ni = 0; ni < 4; ++ni)
        acc[mi][ni] = wmma_bf16(af[mi], bf4[ni], acc[mi][ni]);

    if (kt + 1 < KT) {
      stageTile(buf ^ 1);
      __syncthreads();
    }
  }

  // --- epilogue: C/D layout lane[3:0]=col, VGPR r = row r + 8*lane[4] ---
  const int colb = bn + wn + (lane & 15);
  const int rowhalf = (lane >> 4) * 8;
  #pragma unroll
  for (int mi = 0; mi < 4; ++mi) {
    #pragma unroll
    for (int ni = 0; ni < 4; ++ni) {
      int col = colb + ni * 16;
      float bval = (flags & F_BIAS) ? bias[col] : 0.f;
      #pragma unroll
      for (int r = 0; r < 8; ++r) {
        int row = bm + wm + mi * 16 + rowhalf + r;
        float v = acc[mi][ni][r] + bval;
        if (flags & F_RELU) v = fmaxf(v, 0.f);
        if (flags & F_SCATTER) {
          atomicAdd(&C[(size_t)sidx[row] * N + col], v * srow[row]);
        } else if (flags & F_ACCUM) {
          C[(size_t)row * N + col] += v;
        } else {
          C[(size_t)row * N + col] = v;
        }
      }
    }
  }
}

// ---------------------------------------------------------------------------
// Flash attention, hd=64, non-causal.  grid = (B*H, T/64), 128 threads (4
// waves).  Each wave owns 16 query rows; S = Q@K^T and O += P@V via WMMA bf16.
// K/V tiles are double-buffered with register-staged prefetch; the softmax
// scale 1/sqrt(64) is folded into the Q bf16 conversion.
// Writes the reference's scrambled (B,H,T,hd)->(B,C,T)->(B,T,C) permutation.
// ---------------------------------------------------------------------------
#define LQT 72   // 64 cols + 8 pad (144B pitch, 16B aligned)

__launch_bounds__(128)
__global__ void attn_kernel(const float* __restrict__ qkv, float* __restrict__ out)
{
  const int bh = blockIdx.x, b = bh / Hn, h = bh % Hn;
  const int q0 = blockIdx.y * 64;
  const int tid = threadIdx.x, lane = tid & 31, wid = tid >> 5;
  const int rowhalf = (lane >> 4) * 8;

  __shared__ __align__(16) unsigned short Qs[64 * LQT];
  __shared__ __align__(16) unsigned short Ks[2][64 * LQT];   // Ks[key][d]
  __shared__ __align__(16) unsigned short Vs[2][64 * LQT];   // Vs[d][key]
  __shared__ __align__(16) unsigned short Ps[4][16 * LQT];

  float4 kva[4], vva[4];
  auto loadKV = [&](int j0){
    #pragma unroll
    for (int i = 0; i < 4; ++i) {
      int q = tid + 128 * i;
      int r = q >> 3, c4 = (q & 7) * 4;
      size_t base = (size_t)(b * Tsz + j0 + r) * 2304 + h * 64 + c4;
      kva[i] = *(const float4*)(qkv + base + 768);
      vva[i] = *(const float4*)(qkv + base + 1536);
    }
  };
  auto stageKV = [&](int buf){
    #pragma unroll
    for (int i = 0; i < 4; ++i) {
      int q = tid + 128 * i;
      int r = q >> 3, c4 = (q & 7) * 4;
      unsigned short* d = &Ks[buf][r * LQT + c4];
      d[0] = f2bf(kva[i].x); d[1] = f2bf(kva[i].y); d[2] = f2bf(kva[i].z); d[3] = f2bf(kva[i].w);
      Vs[buf][(c4 + 0) * LQT + r] = f2bf(vva[i].x);
      Vs[buf][(c4 + 1) * LQT + r] = f2bf(vva[i].y);
      Vs[buf][(c4 + 2) * LQT + r] = f2bf(vva[i].z);
      Vs[buf][(c4 + 3) * LQT + r] = f2bf(vva[i].w);
    }
  };

  // stage Q tile (64 rows x 64 d) as bf16, pre-scaled by 1/sqrt(hd)
  #pragma unroll
  for (int i = 0; i < 4; ++i) {
    int q = tid + 128 * i;
    int r = q >> 3, c4 = (q & 7) * 4;
    float4 v = *(const float4*)(qkv + (size_t)(b * Tsz + q0 + r) * 2304 + h * 64 + c4);
    unsigned short* d = &Qs[r * LQT + c4];
    d[0] = f2bf(v.x * 0.125f); d[1] = f2bf(v.y * 0.125f);
    d[2] = f2bf(v.z * 0.125f); d[3] = f2bf(v.w * 0.125f);
  }
  loadKV(0);
  stageKV(0);
  __syncthreads();

  v16bf qf[2];
  qf[0] = ldfrag<LQT>(Qs, wid * 16, 0,  lane);
  qf[1] = ldfrag<LQT>(Qs, wid * 16, 32, lane);

  float m_i[8], l_i[8];
  v8f o[4];
  #pragma unroll
  for (int r = 0; r < 8; ++r) { m_i[r] = -1e30f; l_i[r] = 0.f; }
  #pragma unroll
  for (int ni = 0; ni < 4; ++ni) o[ni] = (v8f){0,0,0,0,0,0,0,0};

  const int JT = Tsz / 64;
  for (int jt = 0; jt < JT; ++jt) {
    const int buf = jt & 1;
    if (jt + 1 < JT) loadKV((jt + 1) * 64);   // overlap with compute

    // S = Q @ K^T   (16 rows x 64 keys per wave)
    v8f s[4];
    #pragma unroll
    for (int ni = 0; ni < 4; ++ni) s[ni] = (v8f){0,0,0,0,0,0,0,0};
    #pragma unroll
    for (int ks = 0; ks < 2; ++ks)
      #pragma unroll
      for (int ni = 0; ni < 4; ++ni)
        s[ni] = wmma_bf16(qf[ks], ldfrag<LQT>(Ks[buf], ni * 16, ks * 32, lane), s[ni]);

    // online softmax (row reductions across the 16-lane half)
    float alpha[8];
    #pragma unroll
    for (int r = 0; r < 8; ++r) {
      float mx = -1e30f;
      #pragma unroll
      for (int ni = 0; ni < 4; ++ni) mx = fmaxf(mx, s[ni][r]);
      #pragma unroll
      for (int msk = 1; msk < 16; msk <<= 1) mx = fmaxf(mx, __shfl_xor(mx, msk, 32));
      float mnew = fmaxf(m_i[r], mx);
      alpha[r] = __expf(m_i[r] - mnew);
      m_i[r] = mnew;
      float rs = 0.f;
      #pragma unroll
      for (int ni = 0; ni < 4; ++ni) {
        float p = __expf(s[ni][r] - mnew);
        s[ni][r] = p; rs += p;
      }
      #pragma unroll
      for (int msk = 1; msk < 16; msk <<= 1) rs += __shfl_xor(rs, msk, 32);
      l_i[r] = l_i[r] * alpha[r] + rs;
      #pragma unroll
      for (int ni = 0; ni < 4; ++ni) o[ni][r] *= alpha[r];
    }

    // P -> per-wave LDS (bf16), then O += P @ V  (wave-local: DS ops are
    // in-order within a wave, no barrier needed)
    unsigned short* pw = &Ps[wid][0];
    #pragma unroll
    for (int ni = 0; ni < 4; ++ni)
      #pragma unroll
      for (int r = 0; r < 8; ++r)
        pw[(rowhalf + r) * LQT + ni * 16 + (lane & 15)] = f2bf(s[ni][r]);
    #pragma unroll
    for (int ks = 0; ks < 2; ++ks) {
      v16bf pa = ldfrag<LQT>(pw, 0, ks * 32, lane);
      #pragma unroll
      for (int ni = 0; ni < 4; ++ni)
        o[ni] = wmma_bf16(pa, ldfrag<LQT>(Vs[buf], ni * 16, ks * 32, lane), o[ni]);
    }

    if (jt + 1 < JT) {
      stageKV(buf ^ 1);
      __syncthreads();
    }
  }

  // epilogue: scrambled write  f = h*T*64 + t*64 + d ; c = f>>10 ; t2 = f&1023
  #pragma unroll
  for (int ni = 0; ni < 4; ++ni) {
    int d = ni * 16 + (lane & 15);
    #pragma unroll
    for (int r = 0; r < 8; ++r) {
      int t = q0 + wid * 16 + rowhalf + r;
      float val = o[ni][r] / l_i[r];
      int f = h * (Tsz * 64) + t * 64 + d;
      int c = f >> 10, t2 = f & (Tsz - 1);
      out[((size_t)b * Tsz + t2) * Dsz + c] = val;
    }
  }
}

// ---------------------------------------------------------------------------
// Small kernels
// ---------------------------------------------------------------------------
__global__ void embed_kernel(const int* __restrict__ ids, const float* __restrict__ tok,
                             const float* __restrict__ pos, float* __restrict__ x)
{
  int bt = blockIdx.x;
  int t = bt & (Tsz - 1);
  int id = ids[bt];
  for (int d = threadIdx.x; d < Dsz; d += 256)
    x[(size_t)bt * Dsz + d] = tok[(size_t)id * Dsz + d] + pos[t * Dsz + d];
}

__global__ void mean_kernel(const float* __restrict__ x, float* __restrict__ xmean)
{
  int idx = blockIdx.x * 256 + threadIdx.x;       // 0 .. B*D-1
  int b = idx / Dsz, d = idx % Dsz;
  float s = 0.f;
  for (int t = 0; t < Tsz; ++t) s += x[((size_t)b * Tsz + t) * Dsz + d];
  xmean[idx] = s * (1.f / Tsz);
}

__global__ void ebm_kernel(const float* __restrict__ xmean, const float* __restrict__ W1,
                           const float* __restrict__ b1, const float* __restrict__ W2,
                           const float* __restrict__ b2, float* __restrict__ out)
{
  int b = blockIdx.x;
  __shared__ float h1[Dsz];
  __shared__ float red[256];
  const float* xm = xmean + b * Dsz;
  for (int j = threadIdx.x; j < Dsz; j += 256) {
    float s = b1[j];
    for (int d = 0; d < Dsz; ++d) s += xm[d] * W1[(size_t)d * Dsz + j];
    h1[j] = fmaxf(s, 0.f);
  }
  __syncthreads();
  float p = 0.f;
  for (int j = threadIdx.x; j < Dsz; j += 256) p += h1[j] * W2[j];
  red[threadIdx.x] = p;
  __syncthreads();
  for (int s = 128; s > 0; s >>= 1) {
    if (threadIdx.x < s) red[threadIdx.x] += red[threadIdx.x + s];
    __syncthreads();
  }
  if (threadIdx.x == 0) out[b * 2 + 1] = red[0] + b2[0];
}

__device__ __forceinline__ float block_sum256(float v, float* sm){
  int lane = threadIdx.x & 31, wid = threadIdx.x >> 5;
  #pragma unroll
  for (int m = 16; m > 0; m >>= 1) v += __shfl_xor(v, m, 32);
  if (lane == 0) sm[wid] = v;
  __syncthreads();
  float r = 0.f;
  #pragma unroll
  for (int w = 0; w < 8; ++w) r += sm[w];
  __syncthreads();
  return r;
}

__device__ __forceinline__ float block_max256(float v, float* sm){
  int lane = threadIdx.x & 31, wid = threadIdx.x >> 5;
  #pragma unroll
  for (int m = 16; m > 0; m >>= 1) v = fmaxf(v, __shfl_xor(v, m, 32));
  if (lane == 0) sm[wid] = v;
  __syncthreads();
  float r = -1e30f;
  #pragma unroll
  for (int w = 0; w < 8; ++w) r = fmaxf(r, sm[w]);
  __syncthreads();
  return r;
}

__launch_bounds__(256)
__global__ void ln_kernel(const float* __restrict__ in, float* __restrict__ outp,
                          const float* __restrict__ g, const float* __restrict__ bb)
{
  __shared__ float sm[8];
  int row = blockIdx.x, tid = threadIdx.x;
  const float* xr = in + (size_t)row * Dsz;
  float v[3], s = 0.f;
  #pragma unroll
  for (int i = 0; i < 3; ++i) { v[i] = xr[tid + 256 * i]; s += v[i]; }
  s = block_sum256(s, sm);
  float mu = s * (1.f / Dsz);
  float q = 0.f;
  #pragma unroll
  for (int i = 0; i < 3; ++i) { float d = v[i] - mu; q += d * d; }
  q = block_sum256(q, sm);
  float rstd = rsqrtf(q * (1.f / Dsz) + 1e-5f);
  #pragma unroll
  for (int i = 0; i < 3; ++i) {
    int d = tid + 256 * i;
    outp[(size_t)row * Dsz + d] = (v[i] - mu) * rstd * g[d] + bb[d];
  }
}

// Router: 8 waves/block, one token per wave.  Lane 0 does noisy top-2 + gates.
__launch_bounds__(256)
__global__ void router_kernel(const float* __restrict__ h, const float* __restrict__ Wr,
                              const float* __restrict__ br, const float* __restrict__ Wn,
                              const float* __restrict__ bn, const float* __restrict__ noise,
                              int* __restrict__ tkidx, float* __restrict__ tkgate)
{
  int tid = threadIdx.x, lane = tid & 31, wid = tid >> 5;
  int t = blockIdx.x * 8 + wid;
  const float* xr = h + (size_t)t * Dsz;
  float xv[24];
  #pragma unroll
  for (int i = 0; i < 24; ++i) xv[i] = xr[lane + 32 * i];
  float rl[En], nl[En];
  #pragma unroll
  for (int e = 0; e < En; ++e) {
    float s1 = 0.f, s2 = 0.f;
    #pragma unroll
    for (int i = 0; i < 24; ++i) {
      int d = lane + 32 * i;
      s1 += xv[i] * Wr[d * En + e];
      s2 += xv[i] * Wn[d * En + e];
    }
    #pragma unroll
    for (int m = 16; m > 0; m >>= 1) { s1 += __shfl_xor(s1, m, 32); s2 += __shfl_xor(s2, m, 32); }
    rl[e] = s1; nl[e] = s2;
  }
  if (lane == 0) {
    float nz[En];
    #pragma unroll
    for (int e = 0; e < En; ++e) {
      float sp = nl[e] + bn[e];
      sp = (sp > 20.f) ? sp : log1pf(__expf(sp));   // softplus
      nz[e] = rl[e] + br[e] + noise[(size_t)t * En + e] * sp;
    }
    int i0 = 0; float v0 = nz[0];
    #pragma unroll
    for (int e = 1; e < En; ++e) if (nz[e] > v0) { v0 = nz[e]; i0 = e; }
    int i1 = -1; float v1 = -1e30f;
    #pragma unroll
    for (int e = 0; e < En; ++e) if (e != i0 && nz[e] > v1) { v1 = nz[e]; i1 = e; }
    float p0 = 1.f / (1.f + __expf(v1 - v0));
    tkidx[t * 2] = i0; tkidx[t * 2 + 1] = i1;
    tkgate[t * 2] = p0; tkgate[t * 2 + 1] = 1.f - p0;
  }
}

// Expert assignment: one block per expert; order-preserving scan over tokens.
__launch_bounds__(1024)
__global__ void assign_kernel(const int* __restrict__ tkidx, const float* __restrict__ tkgate,
                              int* __restrict__ sel, float* __restrict__ selw)
{
  int e = blockIdx.x;
  __shared__ int wsum[32], wpre[32];
  __shared__ int base;
  int tid = threadIdx.x, lane = tid & 31, wid = tid >> 5;
  if (tid == 0) base = 0;
  __syncthreads();
  for (int c = 0; c < Ntok; c += 1024) {
    int t = c + tid;
    int i0 = tkidx[t * 2], i1 = tkidx[t * 2 + 1];
    bool m = (i0 == e) || (i1 == e);
    unsigned bl = (unsigned)__ballot(m);
    int lpre = __popc(bl & ((1u << lane) - 1u));
    if (lane == 0) wsum[wid] = __popc(bl);
    __syncthreads();
    if (tid == 0) {
      int acc = base;
      for (int w = 0; w < 32; ++w) { wpre[w] = acc; acc += wsum[w]; }
      base = acc;
    }
    __syncthreads();
    if (m) {
      int pos = wpre[wid] + lpre;
      if (pos < CAPn) {
        sel[e * CAPn + pos] = t;
        selw[e * CAPn + pos] = (i0 == e) ? tkgate[t * 2] : tkgate[t * 2 + 1];
      }
    }
    __syncthreads();
  }
  int tot = base;
  for (int p = tot + tid; p < CAPn; p += 1024) {   // pad: zero-weight slots
    sel[e * CAPn + p] = 0;
    selw[e * CAPn + p] = 0.f;
  }
}

// Final attention-pooling + regression head.  pooled@Wreg = sum_t w_t (x_t . Wreg)
__launch_bounds__(256)
__global__ void pool_kernel(const float* __restrict__ xf, const float* __restrict__ Wpool,
                            const float* __restrict__ bpool, const float* __restrict__ Wreg,
                            const float* __restrict__ breg, float* __restrict__ out)
{
  int b = blockIdx.x;
  __shared__ float sw[Tsz];
  __shared__ float sm[8];
  const float* xb = xf + (size_t)b * Tsz * Dsz;
  for (int t = threadIdx.x; t < Tsz; t += 256) {
    const float* xr = xb + (size_t)t * Dsz;
    float s = bpool[0];
    for (int d = 0; d < Dsz; ++d) s += xr[d] * Wpool[d];
    sw[t] = s;
  }
  __syncthreads();
  float mx = -1e30f;
  for (int t = threadIdx.x; t < Tsz; t += 256) mx = fmaxf(mx, sw[t]);
  mx = block_max256(mx, sm);
  float ssum = 0.f;
  for (int t = threadIdx.x; t < Tsz; t += 256) {
    float ev = __expf(sw[t] - mx);
    sw[t] = ev; ssum += ev;
  }
  ssum = block_sum256(ssum, sm);
  __syncthreads();
  float acc = 0.f;
  for (int t = threadIdx.x; t < Tsz; t += 256) {
    const float* xr = xb + (size_t)t * Dsz;
    float dot = 0.f;
    for (int d = 0; d < Dsz; ++d) dot += xr[d] * Wreg[d];
    acc += sw[t] * dot;
  }
  acc = block_sum256(acc, sm);
  if (threadIdx.x == 0) out[b * 2 + 0] = acc / ssum + breg[0];
}

// ---------------------------------------------------------------------------
extern "C" void kernel_launch(void* const* d_in, const int* in_sizes, int n_in,
                              void* d_out, int out_size, void* d_ws, size_t ws_size,
                              hipStream_t stream)
{
  (void)in_sizes; (void)n_in; (void)out_size; (void)ws_size;
  const int*   ids   = (const int*)  d_in[0];
  const float* tok   = (const float*)d_in[1];
  const float* pos   = (const float*)d_in[2];
  const float* ln1g  = (const float*)d_in[3];
  const float* ln1b  = (const float*)d_in[4];
  const float* ln2g  = (const float*)d_in[5];
  const float* ln2b  = (const float*)d_in[6];
  const float* Wqkv  = (const float*)d_in[7];
  const float* Wo    = (const float*)d_in[8];
  const float* Wr    = (const float*)d_in[9];
  const float* br    = (const float*)d_in[10];
  const float* Wn    = (const float*)d_in[11];
  const float* bn    = (const float*)d_in[12];
  const float* We1   = (const float*)d_in[13];
  const float* be1   = (const float*)d_in[14];
  const float* We2   = (const float*)d_in[15];
  const float* be2   = (const float*)d_in[16];
  const float* lnfg  = (const float*)d_in[17];
  const float* lnfb  = (const float*)d_in[18];
  const float* Wpool = (const float*)d_in[19];
  const float* bpool = (const float*)d_in[20];
  const float* Wreg  = (const float*)d_in[21];
  const float* breg  = (const float*)d_in[22];
  const float* Webm1 = (const float*)d_in[23];
  const float* bebm1 = (const float*)d_in[24];
  const float* Webm2 = (const float*)d_in[25];
  const float* bebm2 = (const float*)d_in[26];
  const float* noise = (const float*)d_in[27];
  float* out = (float*)d_out;

  char* wsb = (char*)d_ws;
  size_t off = 0;
  auto bump = [&](size_t bytes) { char* p = wsb + off; off += (bytes + 255) & ~(size_t)255; return p; };
  float* x      = (float*)bump((size_t)Ntok * Dsz * 4);
  float* hbuf   = (float*)bump((size_t)Ntok * Dsz * 4);
  float* qkv    = (float*)bump((size_t)Ntok * 3 * Dsz * 4);
  float* attn   = (float*)bump((size_t)Ntok * Dsz * 4);
  float* hid    = (float*)bump((size_t)CAPn * 4 * Dsz * 4);
  int*   tkidx  = (int*)  bump((size_t)Ntok * 2 * 4);
  float* tkgate = (float*)bump((size_t)Ntok * 2 * 4);
  int*   sel    = (int*)  bump((size_t)En * CAPn * 4);
  float* selw   = (float*)bump((size_t)En * CAPn * 4);
  float* xmean  = (float*)bump((size_t)Bsz * Dsz * 4);

  // Embedding + EBM head
  embed_kernel<<<Ntok, 256, 0, stream>>>(ids, tok, pos, x);
  mean_kernel<<<(Bsz * Dsz) / 256, 256, 0, stream>>>(x, xmean);
  ebm_kernel<<<Bsz, 256, 0, stream>>>(xmean, Webm1, bebm1, Webm2, bebm2, out);

  for (int l = 0; l < Ln; ++l) {
    // --- attention block ---
    ln_kernel<<<Ntok, 256, 0, stream>>>(x, hbuf, ln1g + l * Dsz, ln1b + l * Dsz);
    gemm_bf16<<<dim3((3 * Dsz) / BN, Ntok / BM), 256, 0, stream>>>(
        hbuf, Wqkv + (size_t)l * Dsz * 3 * Dsz, nullptr, qkv,
        Ntok, 3 * Dsz, Dsz, nullptr, nullptr, nullptr, 0);
    attn_kernel<<<dim3(Bsz * Hn, Tsz / 64), 128, 0, stream>>>(qkv, attn);
    gemm_bf16<<<dim3(Dsz / BN, Ntok / BM), 256, 0, stream>>>(
        attn, Wo + (size_t)l * Dsz * Dsz, nullptr, x,
        Ntok, Dsz, Dsz, nullptr, nullptr, nullptr, F_ACCUM);

    // --- MoE block ---
    ln_kernel<<<Ntok, 256, 0, stream>>>(x, hbuf, ln2g + l * Dsz, ln2b + l * Dsz);
    router_kernel<<<Ntok / 8, 256, 0, stream>>>(
        hbuf, Wr + (size_t)l * Dsz * En, br + l * En,
        Wn + (size_t)l * Dsz * En, bn + l * En,
        noise + (size_t)l * Ntok * En, tkidx, tkgate);
    assign_kernel<<<En, 1024, 0, stream>>>(tkidx, tkgate, sel, selw);
    for (int e = 0; e < En; ++e) {
      size_t le = (size_t)l * En + e;
      gemm_bf16<<<dim3((4 * Dsz) / BN, CAPn / BM), 256, 0, stream>>>(
          hbuf, We1 + le * Dsz * 4 * Dsz, be1 + le * 4 * Dsz, hid,
          CAPn, 4 * Dsz, Dsz, sel + e * CAPn, nullptr, nullptr,
          F_GATHER | F_BIAS | F_RELU);
      gemm_bf16<<<dim3(Dsz / BN, CAPn / BM), 256, 0, stream>>>(
          hid, We2 + le * 4 * Dsz * Dsz, be2 + le * Dsz, x,
          CAPn, Dsz, 4 * Dsz, nullptr, sel + e * CAPn, selw + e * CAPn,
          F_BIAS | F_SCATTER);
    }
  }

  // Final LN + pooled regression head
  ln_kernel<<<Ntok, 256, 0, stream>>>(x, hbuf, lnfg, lnfb);
  pool_kernel<<<Bsz, 256, 0, stream>>>(hbuf, Wpool, bpool, Wreg, breg, out);
}
